// RNNLSTMRepresentation_74062416052790
// MI455X (gfx1250) — matrix operand order repaired and checked by reading, not compiled, not run on hardware
//
#include <hip/hip_runtime.h>

// ---------------------------------------------------------------------------
// 2-layer LSTM (H=32) fused over time, + final 16-wide projection.
// MI455X (gfx1250, wave32):
//  * one workgroup = 2 waves = one batch tile of 16 rows
//  * wave w owns hidden slice [16w,16w+16) -> gate col-tiles {w, w+2, w+4, w+6}
//  * 16x v_wmma_f32_16x16x32_f16 per timestep per wave; weights pre-swizzled
//    into WMMA B-layout and pinned in VGPRs (no per-step reload traffic)
//  * WMMA chains split into 4 independent "input" WMMAs followed by 4
//    accumulating "recurrent" WMMAs so the f16 WMMA->WMMA RAW hazard slots
//    are filled with real work instead of v_nops
//  * biases folded into activations (pre-scaled fma); accumulators seeded
//    from inline zero -> no bias splat registers, inner loop stays <256 VGPRs
//  * h exchanged via LDS as f16 row-major: A rebuild = 2 ds_load_b128/lane
//  * hardware v_tanh_f32; x streamed once (~11us floor @ 23.3 TB/s) with
//    x(t+1) loads pipelined under activations + layer 2
// ---------------------------------------------------------------------------

#define HID 32
#define TSTEPS 512

typedef __attribute__((ext_vector_type(16))) _Float16 v16h;
typedef __attribute__((ext_vector_type(8)))  _Float16 v8h;
typedef __attribute__((ext_vector_type(8)))  float    v8f;

__device__ __forceinline__ v8f wmma16x16x32(v16h a, v16h b, v8f c) {
  return __builtin_amdgcn_wmma_f32_16x16x32_f16(
      false, a, false, b, (short)0, c, false, false);
}

// ---- activations with pre-scaled bias folded in ---------------------------
#if __has_builtin(__builtin_amdgcn_tanhf)
#define SIGB_SCALE 0.5f        // stored bias for sigmoid gates: 0.5*b
#define TANHB_SCALE 1.0f       // stored bias for g gate: b
__device__ __forceinline__ float sig_pre(float x, float sb) {
  // sigmoid(x+b) = 0.5*tanh(0.5x + 0.5b) + 0.5
  return fmaf(0.5f, __builtin_amdgcn_tanhf(fmaf(0.5f, x, sb)), 0.5f);
}
__device__ __forceinline__ float tanh_pre(float x, float tb) {
  return __builtin_amdgcn_tanhf(x + tb);
}
__device__ __forceinline__ float tanh0(float x) {
  return __builtin_amdgcn_tanhf(x);
}
#else
#define SIGB_SCALE -1.44269504f  // stored bias: -log2(e)*b
#define TANHB_SCALE 2.88539008f  // stored bias: 2*log2(e)*b
__device__ __forceinline__ float sig_pre(float x, float sb) {
  return __builtin_amdgcn_rcpf(
      1.0f + __builtin_amdgcn_exp2f(fmaf(-1.44269504f, x, sb)));
}
__device__ __forceinline__ float tanh_pre(float x, float tb) {
  float r = __builtin_amdgcn_rcpf(
      1.0f + __builtin_amdgcn_exp2f(fmaf(2.88539008f, x, tb)));
  return fmaf(-2.0f, r, 1.0f);
}
__device__ __forceinline__ float tanh0(float x) {
  float r = __builtin_amdgcn_rcpf(
      1.0f + __builtin_amdgcn_exp2f(2.88539008f * x));
  return fmaf(-2.0f, r, 1.0f);
}
#endif

// Build 16x32 f16 A operand from a row-major f32 row (global x path).
// 16-bit A layout: lane L holds M=L%16; lanes0-15: elems = K0..7,K16..23;
// lanes16-31: K8..15, K24..31.
__device__ __forceinline__ v16h buildA_f32(const float* __restrict__ row, int koff) {
  v16h a;
#pragma unroll
  for (int e = 0; e < 8; ++e) {
    a[e]     = (_Float16)row[koff + e];
    a[8 + e] = (_Float16)row[16 + koff + e];
  }
  return a;
}

// Build A operand from a row-major f16 row in LDS: two contiguous 16B loads.
__device__ __forceinline__ v16h buildA_f16(const _Float16* row, int koff) {
  v8h lo = *(const v8h*)(row + koff);
  v8h hi = *(const v8h*)(row + 16 + koff);
  v16h a;
#pragma unroll
  for (int e = 0; e < 8; ++e) { a[e] = lo[e]; a[8 + e] = hi[e]; }
  return a;
}

__global__ __launch_bounds__(64) void lstm2_proj_kernel(
    const float* __restrict__ x,
    const float* __restrict__ Wih1, const float* __restrict__ Whh1,
    const float* __restrict__ bih1, const float* __restrict__ bhh1,
    const float* __restrict__ Wih2, const float* __restrict__ Whh2,
    const float* __restrict__ bih2, const float* __restrict__ bhh2,
    const float* __restrict__ Wproj, const float* __restrict__ bproj,
    float* __restrict__ out) {
  const int lane = threadIdx.x & 31;
  const int w    = threadIdx.x >> 5;      // hidden-half owner
  const int r    = lane & 15;
  const int hi   = (lane >> 4) & 1;
  const int koff  = hi ? 8 : 0;           // A-layout K offset
  const int koff2 = hi ? 16 : 0;          // unused; B operands use koff
  const int tileBase = blockIdx.x * 16;

  __shared__ _Float16 h_lds[2][16][HID];  // [layer][row][hidden], f16

  // ---- weights into WMMA B-layout VGPRs (f16), biases pre-scaled ----------
  v16h Bih1[4], Bhh1[4], Bih2[4], Bhh2[4];
  float b1p[4], b2p[4];
#pragma unroll
  for (int j = 0; j < 4; ++j) {
    const int g = 32 * j + 16 * w + r;    // W row ([4H,32] row-major)
#pragma unroll
    for (int e = 0; e < 16; ++e) {
      Bih1[j][e] = (_Float16)Wih1[g * HID + koff + (e & 7) + ((e >> 3) << 4)];
      Bhh1[j][e] = (_Float16)Whh1[g * HID + koff + (e & 7) + ((e >> 3) << 4)];
      Bih2[j][e] = (_Float16)Wih2[g * HID + koff + (e & 7) + ((e >> 3) << 4)];
      Bhh2[j][e] = (_Float16)Whh2[g * HID + koff + (e & 7) + ((e >> 3) << 4)];
    }
    const float sc = (j == 2) ? TANHB_SCALE : SIGB_SCALE;
    b1p[j] = sc * (bih1[g] + bhh1[g]);
    b2p[j] = sc * (bih2[g] + bhh2[g]);
  }

  // ---- state ----
  v16h h1A = {};
  v16h h2A = {};
  v8f  c1  = {};
  v8f  c2  = {};
  const v8f zero = {};

  const float* xrow0 = x + (size_t)(tileBase + r) * (TSTEPS * HID);
  v16h xA = buildA_f32(xrow0, koff);      // t = 0

  for (int t = 0; t < TSTEPS; ++t) {
    // ------------- layer 1 ---------------------------------------------
    v8f acc[4];
    {
      v8f c0[4];
#pragma unroll
      for (int j = 0; j < 4; ++j) c0[j] = wmma16x16x32(xA, Bih1[j], zero);
#pragma unroll
      for (int j = 0; j < 4; ++j) acc[j] = wmma16x16x32(h1A, Bhh1[j], c0[j]);
    }
    // pipeline: fetch x(t+1) while activations + layer 2 run
    const int tn = (t + 1 < TSTEPS) ? (t + 1) : t;
    const float* xrow = xrow0 + (size_t)tn * HID;
    __builtin_prefetch((const void*)(xrow + HID), 0, 0);
    const v16h xA_next = buildA_f32(xrow, koff);

#pragma unroll
    for (int v = 0; v < 8; ++v) {
      const float ig = sig_pre(acc[0][v], b1p[0]);
      const float fg = sig_pre(acc[1][v], b1p[1]);
      const float gg = tanh_pre(acc[2][v], b1p[2]);
      const float og = sig_pre(acc[3][v], b1p[3]);
      const float c  = fmaf(fg, c1[v], ig * gg);
      c1[v] = c;
      h_lds[0][v + 8 * hi][16 * w + r] = (_Float16)(og * tanh0(c));
    }
    __syncthreads();                      // publish h1[t]
    h1A = buildA_f16(&h_lds[0][r][0], koff);

    // ------------- layer 2 ---------------------------------------------
    v8f acc2[4];
    {
      v8f c0[4];
#pragma unroll
      for (int j = 0; j < 4; ++j) c0[j] = wmma16x16x32(h1A, Bih2[j], zero);
#pragma unroll
      for (int j = 0; j < 4; ++j) acc2[j] = wmma16x16x32(h2A, Bhh2[j], c0[j]);
    }
#pragma unroll
    for (int v = 0; v < 8; ++v) {
      const float ig = sig_pre(acc2[0][v], b2p[0]);
      const float fg = sig_pre(acc2[1][v], b2p[1]);
      const float gg = tanh_pre(acc2[2][v], b2p[2]);
      const float og = sig_pre(acc2[3][v], b2p[3]);
      const float c  = fmaf(fg, c2[v], ig * gg);
      c2[v] = c;
      h_lds[1][v + 8 * hi][16 * w + r] = (_Float16)(og * tanh0(c));
    }
    __syncthreads();                      // publish h2[t]
    h2A = buildA_f16(&h_lds[1][r][0], koff);

    xA = xA_next;
  }

  // ------------- projection: out = h2[T-1] @ Wproj^T + bproj --------------
  v16h Bp;
#pragma unroll
  for (int e = 0; e < 16; ++e) Bp[e] = (_Float16)Wproj[r * HID + koff + (e & 7) + ((e >> 3) << 4)];
  const float bp = bproj[r];

  v8f po = wmma16x16x32(h2A, Bp, zero);
  if (threadIdx.x < 32) {
#pragma unroll
    for (int v = 0; v < 8; ++v)
      out[(size_t)(tileBase + v + 8 * hi) * 16 + r] = po[v] + bp;
  }
}

extern "C" void kernel_launch(void* const* d_in, const int* in_sizes, int n_in,
                              void* d_out, int out_size, void* d_ws, size_t ws_size,
                              hipStream_t stream) {
  const float* x     = (const float*)d_in[0];
  const float* Wih1  = (const float*)d_in[1];
  const float* Whh1  = (const float*)d_in[2];
  const float* bih1  = (const float*)d_in[3];
  const float* bhh1  = (const float*)d_in[4];
  const float* Wih2  = (const float*)d_in[5];
  const float* Whh2  = (const float*)d_in[6];
  const float* bih2  = (const float*)d_in[7];
  const float* bhh2  = (const float*)d_in[8];
  const float* Wproj = (const float*)d_in[9];
  const float* bproj = (const float*)d_in[10];
  float* out = (float*)d_out;

  const int B = in_sizes[0] / (TSTEPS * HID);   // 4096
  dim3 grid(B / 16), block(64);                 // 256 blocks x 2 waves
  hipLaunchKernelGGL(lstm2_proj_kernel, grid, block, 0, stream,
                     x, Wih1, Whh1, bih1, bhh1, Wih2, Whh2, bih2, bhh2,
                     Wproj, bproj, out);
}
